// GlobalAttention_46815143526904
// MI455X (gfx1250) — matrix operand order repaired and checked by reading, not compiled, not run on hardware
//
#include <hip/hip_runtime.h>

// Problem constants (match reference)
#define DIM    512
#define NHEAD  8
#define HD     64
#define SEQ    4096
#define BATCH  2
#define ROWS   (BATCH * SEQ)      // 8192
#define QKVCOL (3 * DIM)          // 1536
#define ATT_SCALE 0.125f          // HD^-0.5

typedef __attribute__((ext_vector_type(16))) __bf16 bv16;
typedef __attribute__((ext_vector_type(8)))  __bf16 bv8;
typedef __attribute__((ext_vector_type(8)))  float  fv8;

// ---------------------------------------------------------------------------
// WMMA helper: D = A(16x32 bf16) * B(32x16 bf16) + C(16x16 f32)
// ---------------------------------------------------------------------------
static __device__ __forceinline__ fv8 wmma_bf16(bv16 a, bv16 b, fv8 c) {
  return __builtin_amdgcn_wmma_f32_16x16x32_bf16(false, a, false, b,
                                                 (short)0, c, false, false);
}

// A-fragment (16x32, 16-bit): lane<16 holds row=lane, K elems {0..7,16..23};
// lane>=16 holds row=lane-16, K elems {8..15,24..31}.  base = &row[kc].
static __device__ __forceinline__ bv16 load_a32(const __bf16* base, int half) {
  bv8 lo = *(const bv8*)(base + 8 * half);
  bv8 hi = *(const bv8*)(base + 16 + 8 * half);
  bv16 a;
#pragma unroll
  for (int i = 0; i < 8; ++i) { a[i] = lo[i]; a[i + 8] = hi[i]; }
  return a;
}

// B-fragment (32x16, 16-bit): lane col = lane&15, lane<16 K elems {0..7,16..23}, lane>=16 K elems {8..15,24..31}.
// base = &Bcol[kc] with B stored [col][k] contiguous in k.
static __device__ __forceinline__ bv16 load_b32(const __bf16* base, int half) {
  bv8 lo = *(const bv8*)(base + 8 * half);
  bv8 hi = *(const bv8*)(base + 16 + 8 * half);
  bv16 b;
#pragma unroll
  for (int i = 0; i < 8; ++i) { b[i] = lo[i]; b[i + 8] = hi[i]; }
  return b;
}

// ---------------------------------------------------------------------------
// Prep: fp32 -> bf16 cast and transposed-cast of weights
// ---------------------------------------------------------------------------
__global__ void cast_bf16_kernel(const float* __restrict__ src,
                                 __bf16* __restrict__ dst, int n) {
  int i = blockIdx.x * blockDim.x + threadIdx.x;
  if (i < n) dst[i] = (__bf16)src[i];
}

// src: [n_in][n_out] row-major (x @ W layout) -> dst: [n_out][n_in] bf16
__global__ void transpose_bf16_kernel(const float* __restrict__ src,
                                      __bf16* __restrict__ dst,
                                      int n_in, int n_out) {
  int o = blockIdx.x * 16 + threadIdx.x;
  int i = blockIdx.y * 16 + threadIdx.y;
  if (o < n_out && i < n_in) dst[o * n_in + i] = (__bf16)src[i * n_out + o];
}

// ---------------------------------------------------------------------------
// One double-buffered GEMM k-step: issue loads for k=knext into (an,bn),
// then run 8 WMMAs consuming (ac,bc).  Ping-ponged by the caller (no copies).
// ---------------------------------------------------------------------------
static __device__ __forceinline__ void gemm_step(
    const __bf16* arow0, const __bf16* arow1, const __bf16* bcol0,
    int knext, int half,
    bv16 (&ac)[2], bv16 (&bc)[4],
    bv16 (&an)[2], bv16 (&bn)[4],
    fv8 (&acc)[2][4]) {
  an[0] = load_a32(arow0 + knext, half);
  an[1] = load_a32(arow1 + knext, half);
#pragma unroll
  for (int j = 0; j < 4; ++j) bn[j] = load_b32(bcol0 + j * 16 * DIM + knext, half);
#pragma unroll
  for (int j = 0; j < 4; ++j) {
    acc[0][j] = wmma_bf16(ac[0], bc[j], acc[0][j]);
    acc[1][j] = wmma_bf16(ac[1], bc[j], acc[1][j]);
  }
}

// ---------------------------------------------------------------------------
// QKV GEMM: [8192 x 512] x [512 x 1536], scatters to Q,K (bh,n,d) and V^T (bh,d,n)
// Block: 256 thr = 8 waves (4 row x 2 col); block tile 128x128; wave tile 32x64.
// ---------------------------------------------------------------------------
__global__ __launch_bounds__(256) void qkv_gemm_kernel(
    const __bf16* __restrict__ xb,     // [8192][512]
    const __bf16* __restrict__ wT,     // [1536][512]
    __bf16* __restrict__ Qb,           // [16][4096][64]
    __bf16* __restrict__ Kb,           // [16][4096][64]
    __bf16* __restrict__ Vt) {         // [16][64][4096]
  const int lane = threadIdx.x & 31, wv = threadIdx.x >> 5;
  const int half = lane >> 4, l16 = lane & 15;
  const int rt = wv & 3, ct = wv >> 2;
  const int rowt = blockIdx.x * 128 + rt * 32;  // wave row base (32 rows)
  const int colt = blockIdx.y * 128 + ct * 64;  // wave feature base (64 cols)

  fv8 acc[2][4];
#pragma unroll
  for (int rr = 0; rr < 2; ++rr)
#pragma unroll
    for (int j = 0; j < 4; ++j)
#pragma unroll
      for (int r = 0; r < 8; ++r) acc[rr][j][r] = 0.f;

  const __bf16* arow0 = xb + (rowt + l16) * DIM;
  const __bf16* arow1 = xb + (rowt + 16 + l16) * DIM;
  const __bf16* bcol0 = wT + (colt + l16) * DIM;

  bv16 aA[2], bA[4], aB[2], bB[4];
  aA[0] = load_a32(arow0, half);
  aA[1] = load_a32(arow1, half);
#pragma unroll
  for (int j = 0; j < 4; ++j) bA[j] = load_b32(bcol0 + j * 16 * DIM, half);

  for (int kc = 0; kc < DIM; kc += 64) {
    gemm_step(arow0, arow1, bcol0, kc + 32, half, aA, bA, aB, bB, acc);
    gemm_step(arow0, arow1, bcol0, (kc + 64) & (DIM - 1), half, aB, bB, aA, bA, acc);
  }

  // Epilogue: C layout -> lane holds one col, rows r + 8*half (+16*rr)
  const int bidx = rowt >> 12;               // batch (128-row tiles never straddle)
#pragma unroll
  for (int rr = 0; rr < 2; ++rr) {
    const int nbase = ((rowt + 16 * rr) & (SEQ - 1)) + 8 * half;
#pragma unroll
    for (int j = 0; j < 4; ++j) {
      const int f = colt + 16 * j + l16;     // feature in [0,1536)
      const int t = f >> 9;                  // 0=q 1=k 2=v (uniform per frag)
      const int ff = f & (DIM - 1);
      const int h = ff >> 6, d = ff & (HD - 1);
      const int bh = bidx * NHEAD + h;
      if (t == 0) {
        __bf16* p = Qb + ((bh * SEQ) + nbase) * HD + d;
#pragma unroll
        for (int r = 0; r < 8; ++r) p[r * HD] = (__bf16)acc[rr][j][r];
      } else if (t == 1) {
        __bf16* p = Kb + ((bh * SEQ) + nbase) * HD + d;
#pragma unroll
        for (int r = 0; r < 8; ++r) p[r * HD] = (__bf16)acc[rr][j][r];
      } else {
        bv8 pk;
#pragma unroll
        for (int r = 0; r < 8; ++r) pk[r] = (__bf16)acc[rr][j][r];
        *(bv8*)(Vt + (bh * HD + d) * SEQ + nbase) = pk;  // contiguous along n
      }
    }
  }
}

// ---------------------------------------------------------------------------
// One 32-key flash-attention tile: S = Q K^T (kcur frags), prefetch knext into
// knxt frags, online softmax, P->LDS->A-frag, O += P V.  Ping-ponged caller.
// ---------------------------------------------------------------------------
static __device__ __forceinline__ void attn_tile(
    int k0, int knext, const __bf16* Kbase, const __bf16* Vbase,
    bv16 qa0, bv16 qa1,
    bv16 (&kcur)[2][2], bv16 (&knxt)[2][2],
    fv8 (&o)[4], float (&m)[8], float (&l)[8],
    __bf16 (*pw)[32], int half, int l16) {
  // V fragments for this tile — consumed only after softmax
  bv16 vf[4];
#pragma unroll
  for (int dj = 0; dj < 4; ++dj)
    vf[dj] = load_b32(Vbase + (16 * dj + l16) * SEQ + k0, half);

  // S = Q K^T * scale  (two 16x16 f32 tiles = 32 keys)
  fv8 s[2];
#pragma unroll
  for (int n = 0; n < 2; ++n) {
    fv8 z;
#pragma unroll
    for (int r = 0; r < 8; ++r) z[r] = 0.f;
    fv8 t = wmma_bf16(qa0, kcur[n][0], z);
    t = wmma_bf16(qa1, kcur[n][1], t);
#pragma unroll
    for (int r = 0; r < 8; ++r) s[n][r] = t[r] * ATT_SCALE;
  }

  // issue K fragment loads for the NEXT tile (hidden behind softmax)
#pragma unroll
  for (int n = 0; n < 2; ++n) {
    const __bf16* krow = Kbase + (knext + 16 * n + l16) * HD;
    knxt[n][0] = load_b32(krow, half);
    knxt[n][1] = load_b32(krow + 32, half);
  }

  // row max over 32 keys: local max then 16-lane butterfly
  float mt[8];
#pragma unroll
  for (int r = 0; r < 8; ++r) mt[r] = fmaxf(s[0][r], s[1][r]);
#pragma unroll
  for (int mask = 1; mask < 16; mask <<= 1)
#pragma unroll
    for (int r = 0; r < 8; ++r) mt[r] = fmaxf(mt[r], __shfl_xor(mt[r], mask, 32));

  float corr[8], rs[8];
#pragma unroll
  for (int r = 0; r < 8; ++r) {
    float mn = fmaxf(m[r], mt[r]);
    corr[r] = __expf(m[r] - mn);
    m[r] = mn;
    s[0][r] = __expf(s[0][r] - mn);
    s[1][r] = __expf(s[1][r] - mn);
    rs[r] = s[0][r] + s[1][r];
  }
#pragma unroll
  for (int mask = 1; mask < 16; mask <<= 1)
#pragma unroll
    for (int r = 0; r < 8; ++r) rs[r] += __shfl_xor(rs[r], mask, 32);
#pragma unroll
  for (int r = 0; r < 8; ++r) l[r] = l[r] * corr[r] + rs[r];
#pragma unroll
  for (int d = 0; d < 4; ++d)
#pragma unroll
    for (int r = 0; r < 8; ++r) o[d][r] *= corr[r];

  // P: C-layout f32 -> LDS -> A-layout bf16 (per-wave tile, DS in-order)
#pragma unroll
  for (int n = 0; n < 2; ++n)
#pragma unroll
    for (int r = 0; r < 8; ++r)
      pw[r + 8 * half][16 * n + l16] = (__bf16)s[n][r];
  asm volatile("s_wait_dscnt 0" ::: "memory");
  bv8 plo = *(const bv8*)&pw[l16][8 * half];
  bv8 phi = *(const bv8*)&pw[l16][16 + 8 * half];
  bv16 pa;
#pragma unroll
  for (int i = 0; i < 8; ++i) { pa[i] = plo[i]; pa[i + 8] = phi[i]; }

  // O += P * V  (reduction over 32 keys)
#pragma unroll
  for (int dj = 0; dj < 4; ++dj) o[dj] = wmma_bf16(pa, vf[dj], o[dj]);
}

// ---------------------------------------------------------------------------
// Flash attention: block = 4 waves (128 thr), 64 query rows of one (b,h).
// Per wave: 16 query rows; stream 32-key tiles, ping-pong K double buffer.
// ---------------------------------------------------------------------------
__global__ __launch_bounds__(128) void attn_kernel(
    const __bf16* __restrict__ Qb, const __bf16* __restrict__ Kb,
    const __bf16* __restrict__ Vt, __bf16* __restrict__ aout) {
  __shared__ __align__(128) __bf16 pLds[4][16][32];

  const int lane = threadIdx.x & 31, wv = threadIdx.x >> 5;
  const int half = lane >> 4, l16 = lane & 15;
  const int bh = blockIdx.y, b = bh >> 3, h = bh & 7;
  const int row0 = blockIdx.x * 64 + wv * 16;
  __bf16 (*pw)[32] = pLds[wv];

  // Q A-fragments (16 rows x 64 d) loaded once
  const __bf16* qrow = Qb + ((size_t)bh * SEQ + row0 + l16) * HD;
  const bv16 qa0 = load_a32(qrow, half);
  const bv16 qa1 = load_a32(qrow + 32, half);

  fv8 o[4];
  float m[8], l[8];
#pragma unroll
  for (int d = 0; d < 4; ++d)
#pragma unroll
    for (int r = 0; r < 8; ++r) o[d][r] = 0.f;
#pragma unroll
  for (int r = 0; r < 8; ++r) { m[r] = -1e30f; l[r] = 0.f; }

  const __bf16* Kbase = Kb + (size_t)bh * SEQ * HD;
  const __bf16* Vbase = Vt + (size_t)bh * HD * SEQ;

  // preload K fragments for tile k0 = 0 into buffer A
  bv16 kA[2][2], kB[2][2];
#pragma unroll
  for (int n = 0; n < 2; ++n) {
    const __bf16* krow = Kbase + (16 * n + l16) * HD;
    kA[n][0] = load_b32(krow, half);
    kA[n][1] = load_b32(krow + 32, half);
  }

  for (int k0 = 0; k0 < SEQ; k0 += 64) {
    attn_tile(k0, k0 + 32, Kbase, Vbase, qa0, qa1, kA, kB,
              o, m, l, pw, half, l16);
    attn_tile(k0 + 32, (k0 + 64) & (SEQ - 1), Kbase, Vbase, qa0, qa1, kB, kA,
              o, m, l, pw, half, l16);
  }

  // normalize and store to attn_out (b, n, h*64+d) bf16
  float inv[8];
#pragma unroll
  for (int r = 0; r < 8; ++r) inv[r] = 1.0f / l[r];
#pragma unroll
  for (int dj = 0; dj < 4; ++dj) {
    const int f = h * HD + 16 * dj + l16;
#pragma unroll
    for (int r = 0; r < 8; ++r) {
      const int n = row0 + r + 8 * half;
      aout[((size_t)b * SEQ + n) * DIM + f] = (__bf16)(o[dj][r] * inv[r]);
    }
  }
}

// ---------------------------------------------------------------------------
// Output projection: [8192 x 512] x [512 x 512] + bias -> fp32 out
// Same 128x128 block / 32x64 wave tiling, ping-pong double buffer.
// ---------------------------------------------------------------------------
__global__ __launch_bounds__(256) void proj_gemm_kernel(
    const __bf16* __restrict__ ab,     // [8192][512]
    const __bf16* __restrict__ wT,     // [512][512]
    const float* __restrict__ bias, float* __restrict__ out) {
  const int lane = threadIdx.x & 31, wv = threadIdx.x >> 5;
  const int half = lane >> 4, l16 = lane & 15;
  const int rt = wv & 3, ct = wv >> 2;
  const int rowt = blockIdx.x * 128 + rt * 32;
  const int colt = blockIdx.y * 128 + ct * 64;

  fv8 acc[2][4];
#pragma unroll
  for (int rr = 0; rr < 2; ++rr)
#pragma unroll
    for (int j = 0; j < 4; ++j)
#pragma unroll
      for (int r = 0; r < 8; ++r) acc[rr][j][r] = 0.f;

  const __bf16* arow0 = ab + (rowt + l16) * DIM;
  const __bf16* arow1 = ab + (rowt + 16 + l16) * DIM;
  const __bf16* bcol0 = wT + (colt + l16) * DIM;

  bv16 aA[2], bA[4], aB[2], bB[4];
  aA[0] = load_a32(arow0, half);
  aA[1] = load_a32(arow1, half);
#pragma unroll
  for (int j = 0; j < 4; ++j) bA[j] = load_b32(bcol0 + j * 16 * DIM, half);

  for (int kc = 0; kc < DIM; kc += 64) {
    gemm_step(arow0, arow1, bcol0, kc + 32, half, aA, bA, aB, bB, acc);
    gemm_step(arow0, arow1, bcol0, (kc + 64) & (DIM - 1), half, aB, bB, aA, bA, acc);
  }

#pragma unroll
  for (int rr = 0; rr < 2; ++rr) {
#pragma unroll
    for (int j = 0; j < 4; ++j) {
      const int col = colt + 16 * j + l16;
      const float bb = bias[col];
      float* p = out + (size_t)(rowt + 16 * rr + 8 * half) * DIM + col;
#pragma unroll
      for (int r = 0; r < 8; ++r) p[(size_t)r * DIM] = acc[rr][j][r] + bb;
    }
  }
}

// ---------------------------------------------------------------------------
extern "C" void kernel_launch(void* const* d_in, const int* in_sizes, int n_in,
                              void* d_out, int out_size, void* d_ws, size_t ws_size,
                              hipStream_t stream) {
  const float* x      = (const float*)d_in[0];
  const float* w_qkv  = (const float*)d_in[1];
  const float* w_proj = (const float*)d_in[2];
  const float* b_proj = (const float*)d_in[3];
  float* out = (float*)d_out;

  char* ws = (char*)d_ws;
  __bf16* xb     = (__bf16*)(ws);                      // 8192*512*2      = 8 MiB
  __bf16* wqkvT  = (__bf16*)(ws + 8388608);            // 1536*512*2
  __bf16* wprojT = (__bf16*)(ws + 9961472);            // 512*512*2
  __bf16* Qb     = (__bf16*)(ws + 10485760);           // 16*4096*64*2
  __bf16* Kb     = (__bf16*)(ws + 18874368);
  __bf16* Vt     = (__bf16*)(ws + 27262976);
  __bf16* aout   = (__bf16*)(ws + 35651584);           // ends at ~44 MiB

  const int nx = ROWS * DIM;
  cast_bf16_kernel<<<(nx + 255) / 256, 256, 0, stream>>>(x, xb, nx);
  transpose_bf16_kernel<<<dim3(QKVCOL / 16, DIM / 16), dim3(16, 16), 0, stream>>>(
      w_qkv, wqkvT, DIM, QKVCOL);
  transpose_bf16_kernel<<<dim3(DIM / 16, DIM / 16), dim3(16, 16), 0, stream>>>(
      w_proj, wprojT, DIM, DIM);

  qkv_gemm_kernel<<<dim3(ROWS / 128, QKVCOL / 128), 256, 0, stream>>>(
      xb, wqkvT, Qb, Kb, Vt);

  attn_kernel<<<dim3(SEQ / 64, BATCH * NHEAD), 128, 0, stream>>>(Qb, Kb, Vt, aout);

  proj_gemm_kernel<<<dim3(ROWS / 128, DIM / 128), 256, 0, stream>>>(
      aout, wprojT, b_proj, out);
}
